// EIRNN_66090956751133
// MI455X (gfx1250) — compile-verified, matrix operand order checked by reading, not run on hardware
//
#include <hip/hip_runtime.h>

// ---------------- static problem sizes ----------------
#define H1C 1024
#define HHC 2048          // H = H1 + H2
#define IN_DIM 160
#define TTC 512
#define BBC 64
#define ALPHA1 (10.0f/30.0f)
#define ALPHA2 (10.0f/60.0f)
#define SIGMA  0.1f
#define KC_REC (HHC/32)   // 64 bf16 k-chunks of 32
#define KC_IN  (IN_DIM/32) // 5
#define NBLK   64         // persistent workgroups (each: 32 hidden cols)
#define XPK_PER_T (4 * KC_IN * 32 * 16)   // packed x elems per timestep
#define NHALF 0x2000000u  // T*B*H/2 = 33554432 (fits u32)

typedef __attribute__((ext_vector_type(16))) __bf16 v16bf;
typedef __attribute__((ext_vector_type(8)))  __bf16 v8bf;
typedef __attribute__((ext_vector_type(8)))  float  v8f;

// ---------------- helpers ----------------
__device__ __forceinline__ __bf16 f2bf(float f) {
    union { float f; unsigned u; } v; v.f = f;
    unsigned r = v.u + 0x7FFFu + ((v.u >> 16) & 1u);   // round-to-nearest-even
    union { unsigned short s; __bf16 b; } o; o.s = (unsigned short)(r >> 16);
    return o.b;
}

// fragment-packed layout for WMMA 16-bit operands.  For B (K x N): fragment
// (n_tile, kc) = 32 lanes x 16 bf16 contiguous; lanes 0-15: n = lane,
// K in {0..7,16..23}; lanes 16-31: n = lane-16, K in {8..15,24..31}.
// The A-matrix (M x K) layout is identical with m in place of n.
__device__ __forceinline__ size_t pack_index(int n, int k, int KC) {
    int n_tile = n >> 4, n_lo = n & 15;
    int kc = k >> 5, ko = k & 31;
    int lane, j;
    if      (ko <  8) { lane = n_lo;      j = ko;      }
    else if (ko < 16) { lane = n_lo + 16; j = ko - 8;  }
    else if (ko < 24) { lane = n_lo;      j = ko - 8;  }
    else              { lane = n_lo + 16; j = ko - 16; }
    return ((size_t)(n_tile * KC + kc) * 32 + lane) * 16 + j;
}

// ---------------- JAX threefry2x32 noise (key(42), normal), branchless ----
__device__ __forceinline__ unsigned rotl32(unsigned x, int r) {
    return (x << r) | (x >> (32 - r));
}
__device__ __forceinline__ void tfr(unsigned& x0, unsigned& x1, int r) {
    x0 += x1; x1 = rotl32(x1, r); x1 ^= x0;
}
// hi==0: element pairs with (idx, idx+HALF), take .x
// hi==1: element pairs with (idx-HALF, idx), take .y   (uniform in t)
__device__ __forceinline__ float noise_val(unsigned idx, int hi) {
    const unsigned k0 = 0u, k1 = 42u;
    const unsigned ks2 = k0 ^ k1 ^ 0x1BD11BDAu;
    unsigned c0 = idx - (hi ? NHALF : 0u);
    unsigned c1 = c0 + NHALF;
    unsigned x0 = c0 + k0, x1 = c1 + k1;
    tfr(x0,x1,13); tfr(x0,x1,15); tfr(x0,x1,26); tfr(x0,x1, 6);
    x0 += k1;  x1 += ks2 + 1u;
    tfr(x0,x1,17); tfr(x0,x1,29); tfr(x0,x1,16); tfr(x0,x1,24);
    x0 += ks2; x1 += k0 + 2u;
    tfr(x0,x1,13); tfr(x0,x1,15); tfr(x0,x1,26); tfr(x0,x1, 6);
    x0 += k0;  x1 += k1 + 3u;
    tfr(x0,x1,17); tfr(x0,x1,29); tfr(x0,x1,16); tfr(x0,x1,24);
    x0 += k1;  x1 += ks2 + 4u;
    tfr(x0,x1,13); tfr(x0,x1,15); tfr(x0,x1,26); tfr(x0,x1, 6);
    x0 += ks2; x1 += k0 + 5u;
    unsigned bits = hi ? x1 : x0;
    float f = __uint_as_float((bits >> 9) | 0x3f800000u) - 1.0f;  // [0,1)
    float u = 2.0f * f - 1.0f;                                    // [-1,1)
    u = fmaxf(u, -0.99999994f);
    // branchless erfinv (Giles): evaluate both polys, select
    float w  = -__logf((1.0f - u) * (1.0f + u));
    float wl = w - 2.5f;
    float ws = sqrtf(w) - 3.0f;
    float pl =                2.81022636e-08f;
    pl = fmaf(pl, wl,         3.43273939e-07f);
    pl = fmaf(pl, wl,        -3.52338770e-06f);
    pl = fmaf(pl, wl,        -4.39150654e-06f);
    pl = fmaf(pl, wl,         2.18580870e-04f);
    pl = fmaf(pl, wl,        -1.25372503e-03f);
    pl = fmaf(pl, wl,        -4.17768164e-03f);
    pl = fmaf(pl, wl,         2.46640727e-01f);
    pl = fmaf(pl, wl,         1.50140941e+00f);
    float ph =               -2.00214257e-04f;
    ph = fmaf(ph, ws,         1.00950558e-04f);
    ph = fmaf(ph, ws,         1.34934322e-03f);
    ph = fmaf(ph, ws,        -3.67342844e-03f);
    ph = fmaf(ph, ws,         5.73950773e-03f);
    ph = fmaf(ph, ws,        -7.62246130e-03f);
    ph = fmaf(ph, ws,         9.43887047e-03f);
    ph = fmaf(ph, ws,         1.00167406e+00f);
    ph = fmaf(ph, ws,         2.83297682e+00f);
    float p = (w < 5.0f) ? pl : ph;
    return 1.41421356237f * p * u;
}

// ---------------- prep kernels ----------------
__global__ void prep_wrec(const float* __restrict__ W_rec, __bf16* __restrict__ wp) {
    size_t idx = (size_t)blockIdx.x * blockDim.x + threadIdx.x;
    if (idx >= (size_t)HHC * HHC) return;
    int h = (int)(idx / HHC), k = (int)(idx % HHC);
    float w = W_rec[idx];
    bool pos = (h < H1C) != (k < H1C);            // cross-population -> squared
    wp[pack_index(h, k, KC_REC)] = f2bf(pos ? w * w : w);
}

__global__ void prep_win(const float* __restrict__ W_in, __bf16* __restrict__ wp) {
    size_t idx = (size_t)blockIdx.x * blockDim.x + threadIdx.x;
    if (idx >= (size_t)HHC * IN_DIM) return;
    int h = (int)(idx / IN_DIM), i = (int)(idx % IN_DIM);
    float w = W_in[idx];
    float val = (h < H1C) ? (i < 64 ? w : 0.0f)
                          : ((i >= 64 && i < 128) ? w : 0.0f);
    wp[pack_index(h, i, KC_IN)] = f2bf(val);
}

// pre-quantize inputs into A-fragment order: [t][mt][kc][lane][16] bf16
__global__ void prep_x(const float* __restrict__ x, __bf16* __restrict__ xpk) {
    size_t idx = (size_t)blockIdx.x * blockDim.x + threadIdx.x;
    if (idx >= (size_t)TTC * BBC * IN_DIM) return;
    int t = (int)(idx / (BBC * IN_DIM));
    int r = (int)(idx % (BBC * IN_DIM));
    int b = r / IN_DIM, i = r % IN_DIM;
    xpk[(size_t)t * XPK_PER_T + pack_index(b, i, KC_IN)] = f2bf(x[idx]);
}

__global__ void init_state(float* __restrict__ state, __bf16* __restrict__ outbf,
                           unsigned* __restrict__ counter) {
    int idx = blockIdx.x * blockDim.x + threadIdx.x;
    if (idx >= BBC * HHC) return;
    state[idx] = 0.0f;
    outbf[idx] = f2bf(0.0f);    // ping buffer 0 used by step t=0
    if (idx == 0) *counter = 0u;
}

// ---------------- persistent RNN kernel ----------------
// Grid: NBLK=64 blocks x 128 threads (4 wave32).  Block covers 32 hidden
// columns (2 n-tiles).  Wave w: n-tile = (w>>1), K-half = (w&1).  K-loop is
// partially rolled (8 x 4 chunks) to stay under 256 VGPRs; 2 threefry noise
// values per iteration go to LDS so VALU co-executes with XDL WMMA without
// dynamic register indexing.  Wave pairs reduce partials through LDS; each
// wave epilogues 2 M-tiles.  Steps separated by a device-wide L2 barrier.
__global__ __launch_bounds__(128) void rnn_persistent(
    const float*  __restrict__ b_in,    // [H]
    const float*  __restrict__ b_rec,   // [H]
    const __bf16* __restrict__ wrec_p,  // packed [H/16][64][32][16]
    const __bf16* __restrict__ win_p,   // packed [H/16][5][32][16]
    const __bf16* __restrict__ xpk,     // packed [T][4][5][32][16]
    __bf16*       __restrict__ outbf,   // [2][B*H] ping-pong
    float*        __restrict__ state,   // [B*H]
    float*        __restrict__ out,     // d_out
    unsigned*     __restrict__ counter) // device-wide step barrier
{
    __shared__ float redbuf[4][2][32][8];   // [wave][tile][lane][v]   = 8 KB
    __shared__ float nzbuf[4][16][32];      // [wave][elem][lane]      = 8 KB

    const int tid     = threadIdx.x;
    const int lane    = tid & 31;
    const int wave    = tid >> 5;
    const int khalf   = wave & 1;
    const int ntile   = blockIdx.x * 2 + (wave >> 1);
    const int n0      = ntile * 16;
    const int lane_lo = lane & 15;
    const int sel8    = (lane & 16) ? 8 : 0;         // also the C/D +8 row sel
    const int h       = n0 + lane_lo;                // this lane's hidden index
    const float bias  = b_in[h] + 2.0f * b_rec[h];
    const float alpha = (h < H1C) ? ALPHA1 : ALPHA2;
    const float one_ma = 1.0f - alpha;
    const int kc_begin = khalf * (KC_REC / 2);
    const int mt_base  = khalf * 2;                  // M-tiles this wave keeps

    float* fin_s = out + (size_t)TTC * BBC * HHC;    // final_state
    float* fin_o = fin_s + (size_t)BBC * HHC;        // final_output

    for (int t = 0; t < TTC; ++t) {
        const __bf16* outb_prev = outbf + (size_t)(t & 1) * (BBC * HHC);
        __bf16*       outb_next = outbf + (size_t)((t + 1) & 1) * (BBC * HHC);

        v8f acc[4] = {v8f{}, v8f{}, v8f{}, v8f{}};

        const __bf16* wptr = wrec_p +
            ((size_t)(ntile * KC_REC + kc_begin) * 32 + lane) * 16;
        const __bf16* aptr = outb_prev + (size_t)lane_lo * HHC
                           + kc_begin * 32 + sel8;
        const unsigned nbase = (unsigned)t * (BBC * HHC) + (unsigned)h;
        const int      nhi   = (t >= TTC / 2) ? 1 : 0;   // uniform per step

        // ---- recurrent GEMM (8 x 4 chunks) with interleaved noise ----
#pragma unroll 1
        for (int it = 0; it < 8; ++it) {
#pragma unroll
            for (int q = 0; q < 4; ++q) {
                v16bf bfrag = *(const v16bf*)(wptr + (size_t)q * (32 * 16));
#pragma unroll
                for (int mt = 0; mt < 4; ++mt) {
                    const __bf16* ap = aptr + (size_t)mt * (16 * HHC) + q * 32;
                    v8bf lo = *(const v8bf*)ap;
                    v8bf hi = *(const v8bf*)(ap + 16);
                    v16bf afrag;
#pragma unroll
                    for (int i = 0; i < 8; ++i) { afrag[i] = lo[i]; afrag[8+i] = hi[i]; }
                    acc[mt] = __builtin_amdgcn_wmma_f32_16x16x32_bf16(
                        false, afrag, false, bfrag, (short)0, acc[mt], false, false);
                }
            }
            // two of this wave's 16 noise values (VALU under the WMMAs)
#pragma unroll
            for (int r2 = 0; r2 < 2; ++r2) {
                const int u = it * 2 + r2;
                const int b = (mt_base + (u >> 3)) * 16 + (u & 7) + sel8;
                nzbuf[wave][u][lane] =
                    noise_val(nbase + (unsigned)b * HHC, nhi);
            }
            wptr += 4 * 32 * 16;
            aptr += 4 * 32;
        }

        // ---- input projection (K-half-0 waves; pre-packed bf16 fragments) ----
        if (khalf == 0) {
            const __bf16* xp = xpk + (size_t)t * XPK_PER_T;
#pragma unroll
            for (int kc = 0; kc < KC_IN; ++kc) {
                v16bf bfrag = *(const v16bf*)(win_p +
                    ((size_t)(ntile * KC_IN + kc) * 32 + lane) * 16);
#pragma unroll
                for (int mt = 0; mt < 4; ++mt) {
                    v16bf afrag = *(const v16bf*)(xp +
                        ((size_t)(mt * KC_IN + kc) * 32 + lane) * 16);
                    acc[mt] = __builtin_amdgcn_wmma_f32_16x16x32_bf16(
                        false, afrag, false, bfrag, (short)0, acc[mt], false, false);
                }
            }
        }

        // ---- K-half reduction through LDS: ship the 2 tiles we don't keep ----
#pragma unroll
        for (int i = 0; i < 2; ++i) {
            int mt = (mt_base ^ 2) + i;
#pragma unroll
            for (int v = 0; v < 8; ++v) redbuf[wave][i][lane][v] = acc[mt][v];
        }
        __syncthreads();
        {
            const int partner = wave ^ 1;
#pragma unroll
            for (int i = 0; i < 2; ++i) {
                int mt = mt_base + i;
#pragma unroll
                for (int v = 0; v < 8; ++v) acc[mt][v] += redbuf[partner][i][lane][v];
            }
        }

        // ---- epilogue: leaky integration + noise + relu + stores ----
        float* out_t = out + (size_t)t * (BBC * HHC);
#pragma unroll
        for (int i = 0; i < 2; ++i) {
            int mt = mt_base + i;
#pragma unroll
            for (int v = 0; v < 8; ++v) {
                int b = mt * 16 + v + sel8;                   // C/D layout M rows
                size_t sidx = (size_t)b * HHC + h;
                float total = acc[mt][v] + bias;
                float s = state[sidx] * one_ma + total * alpha;
                s *= (1.0f + SIGMA * nzbuf[wave][i * 8 + v][lane]);
                float o = fmaxf(s, 0.0f);
                state[sidx]     = s;
                outb_next[sidx] = f2bf(o);
                out_t[sidx]     = o;
                if (t == TTC - 1) { fin_s[sidx] = s; fin_o[sidx] = o; }
            }
        }

        // ---- device-wide step barrier (release -> signal -> spin -> acquire) ----
        __threadfence();          // make this thread's stores device-visible
        __syncthreads();          // all threads in block done + fenced
        if (tid == 0) {
            __hip_atomic_fetch_add(counter, 1u, __ATOMIC_RELEASE,
                                   __HIP_MEMORY_SCOPE_AGENT);
            const unsigned target = (unsigned)(t + 1) * NBLK;
            while (__hip_atomic_load(counter, __ATOMIC_ACQUIRE,
                                     __HIP_MEMORY_SCOPE_AGENT) < target) {
                __builtin_amdgcn_s_sleep(2);
            }
        }
        __syncthreads();          // block released together
        __threadfence();          // acquire: see other blocks' outb/state writes
    }
}

// ---------------- host launcher ----------------
extern "C" void kernel_launch(void* const* d_in, const int* in_sizes, int n_in,
                              void* d_out, int out_size, void* d_ws, size_t ws_size,
                              hipStream_t stream) {
    const float* x     = (const float*)d_in[0];  // [T,B,IN]
    const float* W_in  = (const float*)d_in[1];  // [H,IN]
    const float* b_in  = (const float*)d_in[2];  // [H]
    const float* W_rec = (const float*)d_in[3];  // [H,H]
    const float* b_rec = (const float*)d_in[4];  // [H]
    float* out = (float*)d_out;

    // workspace layout (bytes)
    char* ws = (char*)d_ws;
    const size_t off_wrec = 0;                                    // 8 MB bf16
    const size_t off_win  = off_wrec + (size_t)HHC * HHC * 2;     // 640 KB bf16
    const size_t off_obf  = off_win  + (size_t)HHC * IN_DIM * 2;  // 512 KB bf16 x2
    const size_t off_st   = off_obf  + (size_t)2 * BBC * HHC * 2; // 512 KB f32
    const size_t off_cnt  = off_st   + (size_t)BBC * HHC * 4;     // 256 B slot
    const size_t off_xpk  = off_cnt  + 256;                       // 10 MB bf16
    __bf16*   wrec_p  = (__bf16*)(ws + off_wrec);
    __bf16*   win_p   = (__bf16*)(ws + off_win);
    __bf16*   obf     = (__bf16*)(ws + off_obf);
    float*    state   = (float*)(ws + off_st);
    unsigned* counter = (unsigned*)(ws + off_cnt);
    __bf16*   xpk     = (__bf16*)(ws + off_xpk);

    prep_wrec<<<((size_t)HHC * HHC + 255) / 256, 256, 0, stream>>>(W_rec, wrec_p);
    prep_win <<<((size_t)HHC * IN_DIM + 255) / 256, 256, 0, stream>>>(W_in, win_p);
    prep_x   <<<((size_t)TTC * BBC * IN_DIM + 255) / 256, 256, 0, stream>>>(x, xpk);
    init_state<<<(BBC * HHC + 255) / 256, 256, 0, stream>>>(state, obf, counter);

    rnn_persistent<<<NBLK, 128, 0, stream>>>(b_in, b_rec, wrec_p, win_p, xpk,
                                             obf, state, out, counter);

    (void)in_sizes; (void)n_in; (void)out_size; (void)ws_size;
}
